// Transformer_42588895708019
// MI455X (gfx1250) — matrix operand order repaired and checked by reading, not compiled, not run on hardware
//
#include <hip/hip_runtime.h>
#include <math.h>

typedef __attribute__((ext_vector_type(16))) __bf16 v16bf;
typedef __attribute__((ext_vector_type(8)))  float  v8f;
typedef unsigned int u32x4 __attribute__((ext_vector_type(4)));
typedef int          s32x4 __attribute__((ext_vector_type(4)));
typedef int          s32x8 __attribute__((ext_vector_type(8)));

#define DIMM   768
#define NHEAD  12
#define DHEAD  64
#define MLPD   3072
#define BATCH  128
#define NTOK   65
#define MROWS  (BATCH*NTOK)     /* 8320 */
#define QKVW   2304
#define NEGBIG (-3.4028234663852886e38f)

// ---------------------------------------------------------------------------
// helpers
// ---------------------------------------------------------------------------
__device__ __forceinline__ v16bf ldfrag(const __bf16* p0, const __bf16* p1) {
    v16bf r;
#pragma unroll
    for (int i = 0; i < 8; ++i) { r[i] = p0[i]; r[i + 8] = p1[i]; }
    return r;
}

__device__ __forceinline__ v8f wmma_bf16(v16bf a, v16bf b, v8f c) {
    return __builtin_amdgcn_wmma_f32_16x16x32_bf16(false, a, false, b, (short)0, c,
                                                   false, false);
}

// CDNA5 async copy: 16B global -> LDS, tracked by ASYNCcnt (ISA 15.18.3 op 98).
// LDS operand is the wave-relative LDS offset == low 32 bits of the flat addr.
__device__ __forceinline__ void async_b128_to_lds(const __bf16* gsrc, __bf16* ldst) {
    unsigned loff = (unsigned)(size_t)ldst;
    unsigned long long ga = (unsigned long long)(size_t)gsrc;
    asm volatile("global_load_async_to_lds_b128 %0, %1, off"
                 :: "v"(loff), "v"(ga) : "memory");
}

__device__ __forceinline__ void wait_async0() {
    asm volatile("s_wait_asynccnt 0x0" ::: "memory");
}

// ---------------------------------------------------------------------------
// CDNA5 Tensor Data Mover: 2D tile (tile_w elems x tile_h rows, 2B elements)
// from a row-major tensor into LDS, optional LDS padding of 16B per 64B row
// (ISA ch.8: D# group0/group1 bitfields; pad_interval=3 -> 16 DWORDs,
//  pad_amount=3 -> 4 DWORDs => LDS row stride 80B = 40 bf16 elements).
// ---------------------------------------------------------------------------
__device__ __forceinline__ void tdm_load_2d_pad(const void* gsrc, void* ldst,
                                                unsigned tile_w, unsigned tile_h,
                                                unsigned tensor_w, unsigned row_stride,
                                                int pad) {
    const unsigned long long ga = (unsigned long long)(size_t)gsrc;
    const unsigned lds = (unsigned)(size_t)ldst;
    u32x4 g0;
    g0.x = 1u;                                        // count=1 (valid user D#)
    g0.y = lds;                                       // lds_addr (bytes)
    g0.z = (unsigned)ga;                              // global_addr[31:0]
    g0.w = (unsigned)((ga >> 32) & 0x01ffffffu) | (2u << 30);  // addr[56:32], type=2
    const unsigned w0 = (1u << 16)                    // data_size = 1 (2 bytes)
                      | (pad ? (1u << 20) : 0u)       // pad_enable
                      | (pad ? (3u << 22) : 0u)       // pad_interval: 16 DWORDs
                      | (pad ? (3u << 25) : 0u);      // pad_amount:   4 DWORDs
    const unsigned td0 = tensor_w;                    // tensor dim0 (elements)
    const unsigned td1 = 0x00100000u;                 // tensor dim1 (rows, large)
    s32x8 g1;
    g1[0] = (int)w0;
    g1[1] = (int)((td0 & 0xffffu) << 16);                             // dim0[15:0]
    g1[2] = (int)(((td0 >> 16) & 0xffffu) | ((td1 & 0xffffu) << 16)); // dim0 hi,dim1 lo
    g1[3] = (int)(((td1 >> 16) & 0xffffu) | ((tile_w & 0xffffu) << 16)); // tile_dim0
    g1[4] = (int)(tile_h & 0xffffu);                  // tile_dim1; tile_dim2=0
    g1[5] = (int)row_stride;                          // tensor_dim0_stride[31:0]
    g1[6] = 0;
    g1[7] = 0;
    const s32x4 gz = {0, 0, 0, 0};
#if __has_include(<hip/amd_detail/amd_gfx1250_TDM.h>)
    const s32x8 gz8 = {0, 0, 0, 0, 0, 0, 0, 0};
    __builtin_amdgcn_tensor_load_to_lds(g0, g1, gz, gz, gz8, 0);
#else
    __builtin_amdgcn_tensor_load_to_lds(g0, g1, gz, gz, 0);
#endif
}

// ---------------------------------------------------------------------------
// fp32 -> bf16 conversion / fp32 copy
// ---------------------------------------------------------------------------
__global__ __launch_bounds__(256) void cvt_kernel(const float* __restrict__ s,
                                                  __bf16* __restrict__ d, size_t n) {
    for (size_t i = (size_t)blockIdx.x * 256 + threadIdx.x; i < n;
         i += (size_t)gridDim.x * 256)
        d[i] = (__bf16)s[i];
}

__global__ __launch_bounds__(256) void copy_kernel(const float* __restrict__ s,
                                                   float* __restrict__ d, size_t n) {
    for (size_t i = (size_t)blockIdx.x * 256 + threadIdx.x; i < n;
         i += (size_t)gridDim.x * 256)
        d[i] = s[i];
}

// ---------------------------------------------------------------------------
// LayerNorm over DIM=768; one row per 256-thread block
// ---------------------------------------------------------------------------
template <int OBF, int OF>
__global__ __launch_bounds__(256) void ln_kernel(const float* __restrict__ x,
                                                 const float* __restrict__ g,
                                                 const float* __restrict__ bta,
                                                 __bf16* __restrict__ obf,
                                                 float* __restrict__ of) {
    const int row  = blockIdx.x;
    const int tid  = threadIdx.x;
    const int lane = tid & 31, wid = tid >> 5;
    const float* xr = x + (size_t)row * DIMM;
    float v0 = xr[tid], v1 = xr[tid + 256], v2 = xr[tid + 512];
    float s = v0 + v1 + v2, s2 = v0 * v0 + v1 * v1 + v2 * v2;
#pragma unroll
    for (int off = 16; off > 0; off >>= 1) {
        s  += __shfl_xor(s,  off, 32);
        s2 += __shfl_xor(s2, off, 32);
    }
    __shared__ float rs[8], rs2[8];
    if (lane == 0) { rs[wid] = s; rs2[wid] = s2; }
    __syncthreads();
    s = 0.f; s2 = 0.f;
#pragma unroll
    for (int w = 0; w < 8; ++w) { s += rs[w]; s2 += rs2[w]; }
    const float mu   = s * (1.f / 768.f);
    const float var  = s2 * (1.f / 768.f) - mu * mu;
    const float rstd = rsqrtf(var + 1e-5f);
    const float vin[3] = {v0, v1, v2};
#pragma unroll
    for (int i = 0; i < 3; ++i) {
        const int col = tid + 256 * i;
        const float o = (vin[i] - mu) * rstd * g[col] + bta[col];
        if (OBF) obf[(size_t)row * DIMM + col] = (__bf16)o;
        if (OF)  of[(size_t)row * DIMM + col]  = o;
    }
}

// ---------------------------------------------------------------------------
// GEMM B tile loader: 32x128 transposed through VGPRs (TDM can't transpose).
// ---------------------------------------------------------------------------
__device__ __forceinline__ void gemm_load_B(const __bf16* __restrict__ Bm, int Nn,
                                            int blockN, int kt, int tid,
                                            __bf16 (*Bst)[40]) {
#pragma unroll
    for (int c = 0; c < 2; ++c) {
        const int lin  = tid + c * 256;
        const int brow = lin >> 4;
        const int bc   = (lin & 15) * 8;
        uint4 t = *(const uint4*)(Bm + (size_t)(kt + brow) * Nn + blockN + bc);
        const __bf16* tp = (const __bf16*)&t;
#pragma unroll
        for (int e = 0; e < 8; ++e) Bst[bc + e][brow] = tp[e];
    }
}

// ---------------------------------------------------------------------------
// bf16 GEMM: C(MxN) = A(MxK) * B(KxN). Double-buffered LDS; A tile streamed by
// the Tensor Data Mover (wave 0 issues, TENSORcnt waited by all); B via VGPRs.
// BM=BN=128, BK=32. 256 threads = 8 waves (4x2); each wave: 32x64 (2x4 WMMA).
// ---------------------------------------------------------------------------
template <int HAS_BIAS, int HAS_RES, int HAS_OUT, int ACT>
__global__ __launch_bounds__(256, 1) void gemm_bf16_kernel(
    const __bf16* __restrict__ A, const __bf16* __restrict__ Bm,
    const float* __restrict__ bias, __bf16* __restrict__ outbf,
    float* __restrict__ res, int Nn, int Kd) {
    __shared__ __bf16 As[2][128][40];   // TDM-padded: 64B row + 16B pad = 80B stride
    __shared__ __bf16 Bst[2][128][40];  // transposed: Bst[n][k]

    const int tid  = threadIdx.x;
    const int lane = tid & 31, wid = tid >> 5;
    const int wr = wid >> 1, wc = wid & 1;
    const int r = lane & 15, khalf = lane >> 4;
    const int blockM = blockIdx.y * 128;
    const int blockN = blockIdx.x * 128;

    const v8f vz = {0.f, 0.f, 0.f, 0.f, 0.f, 0.f, 0.f, 0.f};
    v8f acc[2][4];
#pragma unroll
    for (int mi = 0; mi < 2; ++mi)
#pragma unroll
        for (int ni = 0; ni < 4; ++ni) acc[mi][ni] = vz;

    if (wid == 0)
        tdm_load_2d_pad(A + (size_t)blockM * Kd, &As[0][0][0], 32, 128, Kd, Kd, 1);
    gemm_load_B(Bm, Nn, blockN, 0, tid, Bst[0]);
    __builtin_amdgcn_s_wait_tensorcnt(0);
    __syncthreads();

    int cur = 0;
    for (int kt = 0; kt < Kd; kt += 32) {
        if (kt + 32 < Kd) {
            if (wid == 0)
                tdm_load_2d_pad(A + (size_t)blockM * Kd + kt + 32,
                                &As[cur ^ 1][0][0], 32, 128, Kd, Kd, 1);
            gemm_load_B(Bm, Nn, blockN, kt + 32, tid, Bst[cur ^ 1]);
        }
        if (kt + 64 < Kd)  // stream weights ahead into GL2
            __builtin_prefetch(Bm + (size_t)(kt + 64 + (tid >> 4)) * Nn + blockN +
                                   (tid & 15) * 8, 0, 1);

        v16bf af[2], bfr[4];
#pragma unroll
        for (int mi = 0; mi < 2; ++mi) {
            const __bf16* p = &As[cur][wr * 32 + mi * 16 + r][khalf * 16];
            af[mi] = ldfrag(p, p + 8);
        }
#pragma unroll
        for (int ni = 0; ni < 4; ++ni) {
            const __bf16* p = &Bst[cur][wc * 64 + ni * 16 + r][khalf * 16];
            bfr[ni] = ldfrag(p, p + 8);
        }
#pragma unroll
        for (int mi = 0; mi < 2; ++mi)
#pragma unroll
            for (int ni = 0; ni < 4; ++ni)
                acc[mi][ni] = wmma_bf16(af[mi], bfr[ni], acc[mi][ni]);

        __builtin_amdgcn_s_wait_tensorcnt(0);
        __syncthreads();
        cur ^= 1;
    }

    // ---- epilogue (fully specialized; no runtime branches) ----
#pragma unroll
    for (int mi = 0; mi < 2; ++mi) {
#pragma unroll
        for (int ni = 0; ni < 4; ++ni) {
            const int colg = blockN + wc * 64 + ni * 16 + r;
            const int row0 = blockM + wr * 32 + mi * 16 + 8 * khalf;
            const float bia = HAS_BIAS ? bias[colg] : 0.f;
#pragma unroll
            for (int v = 0; v < 8; ++v) {
                float val = acc[mi][ni][v] + bia;
                if (ACT) val = 0.5f * val * (1.f + erff(val * 0.70710678118654752f));
                const size_t idx = (size_t)(row0 + v) * Nn + colg;
                if (HAS_RES) res[idx] += val;
                if (HAS_OUT) outbf[idx] = (__bf16)val;
            }
        }
    }
}

// ---------------------------------------------------------------------------
// Attention: one block per (b,h). scores = (q k^T)*scale + relbias;
// exact top-16 per row; masked softmax; out = P @ v. All matmuls via WMMA.
// ---------------------------------------------------------------------------
__global__ __launch_bounds__(256, 1) void attn_kernel(
    const __bf16* __restrict__ qkv, const float* __restrict__ relb,
    const int* __restrict__ rpi, __bf16* __restrict__ outb) {
    __shared__ float  sS[80][81];       // scores (padded to 80 rows)
    __shared__ __bf16 vSt[64][104];     // v transposed: vSt[d][n], n padded to 96
    __shared__ __bf16 qkbuf[11520];     // q/k in phase 1; reused as P afterwards
    __bf16 (*qS)[72]  = (__bf16(*)[72])qkbuf;
    __bf16 (*kS)[72]  = (__bf16(*)[72])(qkbuf + 80 * 72);
    __bf16 (*pS)[104] = (__bf16(*)[104])qkbuf;   // 80*104 = 8320 <= 11520

    const int tid  = threadIdx.x;
    const int lane = tid & 31, wid = tid >> 5;
    const int bh = blockIdx.x, bb = bh / NHEAD, hh = bh % NHEAD;
    const __bf16* qg = qkv + (size_t)bb * NTOK * QKVW + hh * DHEAD;

    // ---- q,k rows 0..64 streamed via async-to-LDS (16B chunks) ----
    for (int i = tid; i < NTOK * 8; i += 256) {
        const int n = i >> 3, c8 = (i & 7) * 8;
        async_b128_to_lds(qg + (size_t)n * QKVW + c8, &qS[n][c8]);
        async_b128_to_lds(qg + (size_t)n * QKVW + 768 + c8, &kS[n][c8]);
    }
    // ---- zero the padded rows 65..79 ----
    for (int i = tid; i < 15 * 64; i += 256) {
        const int n = 65 + (i >> 6), dc = i & 63;
        qS[n][dc] = (__bf16)0.f;
        kS[n][dc] = (__bf16)0.f;
    }
    // ---- v transposed (64 x 96, zero-padded cols) ----
    for (int i = tid; i < 64 * 96; i += 256) {
        const int dc = i / 96, n = i % 96;
        __bf16 vv = (__bf16)0.f;
        if (n < NTOK) vv = qg[(size_t)n * QKVW + 1536 + dc];
        vSt[dc][n] = vv;
    }
    wait_async0();
    __syncthreads();

    const int r = lane & 15, khalf = lane >> 4;
    const v8f vz = {0.f, 0.f, 0.f, 0.f, 0.f, 0.f, 0.f, 0.f};

    // ---- phase 1: scores = q k^T * scale, 5x5 tiles of 16x16 ----
    for (int t = wid; t < 25; t += 8) {
        const int tm = t / 5, tn = t % 5;
        v8f acc = vz;
#pragma unroll
        for (int kk = 0; kk < 64; kk += 32) {
            const __bf16* ap = &qS[tm * 16 + r][kk + khalf * 16];
            const v16bf a = ldfrag(ap, ap + 8);
            const __bf16* bp = &kS[tn * 16 + r][kk + khalf * 16];  // B = k^T
            const v16bf b = ldfrag(bp, bp + 8);
            acc = wmma_bf16(a, b, acc);
        }
#pragma unroll
        for (int v = 0; v < 8; ++v)
            sS[tm * 16 + 8 * khalf + v][tn * 16 + r] = acc[v] * 0.125f;  // SCALE
    }
    __syncthreads();

    // ---- phase 2: relative position bias on (1..64, 1..64) ----
    for (int idx = tid; idx < 64 * 64; idx += 256) {
        const int i = (idx >> 6) + 1, j = (idx & 63) + 1;
        sS[i][j] += relb[rpi[idx]];
    }
    __syncthreads();

    // ---- phase 3: exact top-16 per row + masked softmax -> pS (bf16) ----
    for (int row = wid; row < 80; row += 8) {
        if (row >= NTOK) {
#pragma unroll
            for (int s = 0; s < 3; ++s) {
                const int j = lane + 32 * s;
                if (j < 96) pS[row][j] = (__bf16)0.f;
            }
            continue;
        }
        float vals[3]; bool cl[3];
#pragma unroll
        for (int s = 0; s < 3; ++s) {
            const int j = lane + 32 * s;
            vals[s] = (j < NTOK) ? sS[row][j] : NEGBIG;
            cl[s] = false;
        }
        float rowmax = NEGBIG;
        for (int it = 0; it < 16; ++it) {
            float bv = NEGBIG; int bi = 1 << 20;
#pragma unroll
            for (int s = 0; s < 3; ++s)
                if (!cl[s] && vals[s] > bv) { bv = vals[s]; bi = lane + 32 * s; }
#pragma unroll
            for (int off = 16; off > 0; off >>= 1) {
                const float ov = __shfl_xor(bv, off, 32);
                const int   oi = __shfl_xor(bi, off, 32);
                if (ov > bv || (ov == bv && oi < bi)) { bv = ov; bi = oi; }
            }
            if (it == 0) rowmax = bv;
            if ((bi & 31) == lane && (bi >> 5) < 3) cl[bi >> 5] = true;
        }
        float p[3], lsum = 0.f;
#pragma unroll
        for (int s = 0; s < 3; ++s) {
            p[s] = cl[s] ? expf(vals[s] - rowmax) : 0.f;
            lsum += p[s];
        }
#pragma unroll
        for (int off = 16; off > 0; off >>= 1) lsum += __shfl_xor(lsum, off, 32);
        const float inv = 1.f / lsum;
#pragma unroll
        for (int s = 0; s < 3; ++s) {
            const int j = lane + 32 * s;
            if (j < 96) pS[row][j] = (__bf16)(p[s] * inv);
        }
    }
    __syncthreads();

    // ---- phase 4: out = P(65x96) @ V(96x64), 5x4 tiles ----
    for (int t = wid; t < 20; t += 8) {
        const int tm = t / 4, tn = t % 4;
        v8f acc = vz;
#pragma unroll
        for (int kk = 0; kk < 96; kk += 32) {
            const __bf16* ap = &pS[tm * 16 + r][kk + khalf * 16];
            const v16bf a = ldfrag(ap, ap + 8);
            const __bf16* bp = &vSt[tn * 16 + r][kk + khalf * 16];
            const v16bf b = ldfrag(bp, bp + 8);
            acc = wmma_bf16(a, b, acc);
        }
#pragma unroll
        for (int v = 0; v < 8; ++v) {
            const int rowg = tm * 16 + 8 * khalf + v;
            if (rowg < NTOK)
                outb[((size_t)bb * NTOK + rowg) * DIMM + hh * DHEAD + tn * 16 + r] =
                    (__bf16)acc[v];
        }
    }
}

// ---------------------------------------------------------------------------
// host orchestration
// ---------------------------------------------------------------------------
extern "C" void kernel_launch(void* const* d_in, const int* in_sizes, int n_in,
                              void* d_out, int out_size, void* d_ws, size_t ws_size,
                              hipStream_t stream) {
    (void)in_sizes; (void)n_in; (void)out_size; (void)ws_size;
    const float* x_in    = (const float*)d_in[0];
    const float* ln1_g   = (const float*)d_in[1];
    const float* ln1_b   = (const float*)d_in[2];
    const float* w_qkv   = (const float*)d_in[3];
    const float* relbias = (const float*)d_in[4];
    const float* w_out   = (const float*)d_in[5];
    const float* b_out   = (const float*)d_in[6];
    const float* ffln_g  = (const float*)d_in[7];
    const float* ffln_b  = (const float*)d_in[8];
    const float* w1      = (const float*)d_in[9];
    const float* b1      = (const float*)d_in[10];
    const float* w2      = (const float*)d_in[11];
    const float* b2      = (const float*)d_in[12];
    const float* final_g = (const float*)d_in[13];
    const float* final_b = (const float*)d_in[14];
    const int*   rpi     = (const int*)d_in[15];

    size_t off = 0;
    auto alloc = [&](size_t bytes) -> void* {
        void* p = (char*)d_ws + off;
        off += (bytes + 255) & ~(size_t)255;
        return p;
    };
    const size_t E_QKV = (size_t)12 * 768 * 2304;
    const size_t E_OUT = (size_t)12 * 768 * 768;
    const size_t E_W1  = (size_t)12 * 768 * 3072;
    const size_t E_W2  = (size_t)12 * 3072 * 768;

    __bf16* wqkv_bf = (__bf16*)alloc(E_QKV * 2);
    __bf16* wout_bf = (__bf16*)alloc(E_OUT * 2);
    __bf16* w1_bf   = (__bf16*)alloc(E_W1 * 2);
    __bf16* w2_bf   = (__bf16*)alloc(E_W2 * 2);
    float*  xbuf    = (float*)alloc((size_t)MROWS * DIMM * 4);
    __bf16* hbuf    = (__bf16*)alloc((size_t)MROWS * DIMM * 2);
    __bf16* qkvbuf  = (__bf16*)alloc((size_t)MROWS * QKVW * 2);
    __bf16* aobuf   = (__bf16*)alloc((size_t)MROWS * DIMM * 2);
    __bf16* h1buf   = (__bf16*)alloc((size_t)MROWS * MLPD * 2);

    cvt_kernel<<<2048, 256, 0, stream>>>(w_qkv, wqkv_bf, E_QKV);
    cvt_kernel<<<2048, 256, 0, stream>>>(w_out, wout_bf, E_OUT);
    cvt_kernel<<<2048, 256, 0, stream>>>(w1, w1_bf, E_W1);
    cvt_kernel<<<2048, 256, 0, stream>>>(w2, w2_bf, E_W2);
    copy_kernel<<<2048, 256, 0, stream>>>(x_in, xbuf, (size_t)MROWS * DIMM);

    for (int d = 0; d < 12; ++d) {
        ln_kernel<1, 0><<<MROWS, 256, 0, stream>>>(xbuf, ln1_g + d * DIMM,
                                                   ln1_b + d * DIMM, hbuf, nullptr);
        gemm_bf16_kernel<0, 0, 1, 0><<<dim3(QKVW / 128, MROWS / 128), 256, 0, stream>>>(
            hbuf, wqkv_bf + (size_t)d * 768 * 2304, nullptr, qkvbuf, nullptr,
            QKVW, DIMM);
        attn_kernel<<<BATCH * NHEAD, 256, 0, stream>>>(qkvbuf, relbias + d * 225, rpi,
                                                       aobuf);
        gemm_bf16_kernel<1, 1, 0, 0><<<dim3(DIMM / 128, MROWS / 128), 256, 0, stream>>>(
            aobuf, wout_bf + (size_t)d * 768 * 768, b_out + d * DIMM, nullptr, xbuf,
            DIMM, DIMM);
        ln_kernel<1, 0><<<MROWS, 256, 0, stream>>>(xbuf, ffln_g + d * DIMM,
                                                   ffln_b + d * DIMM, hbuf, nullptr);
        gemm_bf16_kernel<1, 0, 1, 1><<<dim3(MLPD / 128, MROWS / 128), 256, 0, stream>>>(
            hbuf, w1_bf + (size_t)d * 768 * 3072, b1 + d * MLPD, h1buf, nullptr,
            MLPD, DIMM);
        gemm_bf16_kernel<1, 1, 0, 0><<<dim3(DIMM / 128, MROWS / 128), 256, 0, stream>>>(
            h1buf, w2_bf + (size_t)d * 3072 * 768, b2 + d * DIMM, nullptr, xbuf,
            DIMM, MLPD);
    }
    ln_kernel<0, 1><<<MROWS, 256, 0, stream>>>(xbuf, final_g, final_b, nullptr,
                                               (float*)d_out);
}
